// EmbeddingPropLayer_4097398800412
// MI455X (gfx1250) — compile-verified, matrix operand order checked by reading, not run on hardware
//
#include <hip/hip_runtime.h>
#include <hip/hip_bf16.h>
#include <math.h>

// ---------------------------------------------------------------------------
// EmbeddingPropagation on MI455X (gfx1250, wave32, WMMA + Tensor Data Mover)
//
// z = (I - alpha*S)^-1 x via 30-term Neumann series (rho(alpha*S) <= 0.5).
// All GEMMs: v_wmma_f32_16x16x32_f16, split-f16 operands (hi+lo planes):
//   a*b ~= ahi*bhi + ahi*blo + alo*bhi   (fp32 accumulate, ~2^-22 rel err)
// Operands are de-interleaved f16 planes; B is always pre-transposed (BT,
// [N][K] row-major) so WMMA fragments are two contiguous ds_load_b128 each.
// Global->LDS staging uses TENSOR_LOAD_TO_LDS (TDM) with double buffering:
// wave 0 issues the next tile's DMA, s_wait_tensorcnt, barrier, compute.
// S (64MB planes) + z (8MB) fit in the 192MB L2 -> iterations compute-bound.
// ---------------------------------------------------------------------------

typedef __attribute__((ext_vector_type(16))) _Float16 v16h;
typedef __attribute__((ext_vector_type(8)))  float    v8f;
typedef __attribute__((ext_vector_type(4)))  unsigned v4u;
typedef __attribute__((ext_vector_type(8)))  int      v8i;
typedef __attribute__((ext_vector_type(4)))  int      v4i;

#if defined(__HIP_DEVICE_COMPILE__) && __has_builtin(__builtin_amdgcn_tensor_load_to_lds)
#define USE_TDM 1
#else
#define USE_TDM 0
#endif

static constexpr int   BN    = 4096;   // batch
static constexpr int   CN    = 512;    // channels
static constexpr int   ON    = 512;    // out features
static constexpr float ALPHA = 0.5f;
static constexpr float RBF   = 1.0f;
static constexpr int   NEUMANN_ITERS = 30;
static constexpr float RSQRT_C = 0.04419417382415922f; // 1/sqrt(512)

// LDS row stride in halfs: 32 data halfs + 4-dword pad (80B rows, 16B aligned)
static constexpr int LDS_STRIDE = 40;

// ---- split-f16 helpers ----------------------------------------------------
__device__ __forceinline__ void split_f32(float v, _Float16& h, _Float16& l) {
    h = (_Float16)v;               // RNE
    l = (_Float16)(v - (float)h);  // residual
}

__device__ __forceinline__ v8f wmma_f16(v16h a, v16h b, v8f c) {
    return __builtin_amdgcn_wmma_f32_16x16x32_f16(
        /*neg_a=*/false, a, /*neg_b=*/false, b,
        /*c_mod=*/(short)0, c, /*reuse_a=*/false, /*reuse_b=*/false);
}

// Fragment: lane's 16 elements are halfs [8h, 8h+8) and [16+8h, 24+8h) of its
// row -> two b128 LDS loads, no shuffles.
__device__ __forceinline__ v16h ld_frag(const _Float16* rowp, int half) {
    struct Q2 { uint4 a, b; } q;
    q.a = *reinterpret_cast<const uint4*>(rowp + (half << 3));
    q.b = *reinterpret_cast<const uint4*>(rowp + 16 + (half << 3));
    return __builtin_bit_cast(v16h, q);
}

#if USE_TDM
// Issue a TDM 2D tile load: rows x 32 halfs, global row stride ldg (halfs),
// into LDS with a 4-dword pad every 16 dwords (-> 40-half LDS row stride).
// D# per CDNA5 ISA ch.8: group0 {count, lds_addr, global_addr, type=2},
// group1 {data_size=2B, pad cfg, tensor/tile dims, dim0 stride}. 2D tensor
// -> remaining groups zero. Tracked by TENSORcnt.
// NOTE: this toolchain exposes the 6-arg builtin form:
//   (uint32x4 g0, int32x8 g1, int32x4, int32x4, int32x8, i32 cpol)
__device__ __forceinline__ void tdm_load_2d(const _Float16* ldsDst,
                                            const _Float16* gsrc,
                                            unsigned rows, unsigned ldg) {
    const unsigned           lds_addr = (unsigned)(size_t)(const void*)ldsDst;
    const unsigned long long ga       = (unsigned long long)(size_t)(const void*)gsrc;
    v4u g0;
    g0[0] = 1u;                                                // count=1, user D#
    g0[1] = lds_addr;                                          // LDS byte address
    g0[2] = (unsigned)ga;                                      // global_addr[31:0]
    g0[3] = (unsigned)((ga >> 32) & 0x01ffffffu) | (2u << 30); // [56:32] + type=2
    const unsigned long long dim0   = 32;                      // halfs per row
    const unsigned long long stride = ldg;
    const unsigned long long q0 = (1ull << 16)   // data_size = 2 bytes
                                | (1ull << 20)   // pad_enable
                                | (3ull << 22)   // pad_interval: 16 dwords
                                | (3ull << 25)   // pad_amount:   4 dwords
                                | (dim0 << 48);  // tensor_dim0[15:0]
    const unsigned long long q1 = (dim0 >> 16)                      // tensor_dim0[31:16]
                                | ((unsigned long long)rows << 16)  // tensor_dim1
                                | (dim0 << 48);                     // tile_dim0
    const unsigned long long q2 = (unsigned long long)rows          // tile_dim1 (tile_dim2=0)
                                | (stride << 32);                   // dim0_stride[31:0]
    const unsigned long long q3 = (stride >> 32);                   // dim0_stride[47:32]
    v8i g1;
    g1[0] = (int)(unsigned)q0; g1[1] = (int)(unsigned)(q0 >> 32);
    g1[2] = (int)(unsigned)q1; g1[3] = (int)(unsigned)(q1 >> 32);
    g1[4] = (int)(unsigned)q2; g1[5] = (int)(unsigned)(q2 >> 32);
    g1[6] = (int)(unsigned)q3; g1[7] = (int)(unsigned)(q3 >> 32);
    const v4i gz4 = {0, 0, 0, 0};
    const v8i gz8 = {0, 0, 0, 0, 0, 0, 0, 0};
    __builtin_amdgcn_tensor_load_to_lds(g0, g1, gz4, gz4, gz8, 0);
}
#endif

// One K-step of the wave micro-kernel: 8 fragment loads + 12 WMMAs.
__device__ __forceinline__ void mma_tile(
    const _Float16* Ah, const _Float16* Al,
    const _Float16* Bh, const _Float16* Bl,
    int waveM, int waveN, int lrow, int half, v8f (&acc)[2][2])
{
    v16h ahi[2], alo[2], bhi[2], blo[2];
    #pragma unroll
    for (int mt = 0; mt < 2; ++mt) {
        const int row = waveM * 32 + mt * 16 + lrow;
        ahi[mt] = ld_frag(Ah + row * LDS_STRIDE, half);
        alo[mt] = ld_frag(Al + row * LDS_STRIDE, half);
    }
    #pragma unroll
    for (int nt = 0; nt < 2; ++nt) {
        const int col = waveN * 32 + nt * 16 + lrow;
        bhi[nt] = ld_frag(Bh + col * LDS_STRIDE, half);
        blo[nt] = ld_frag(Bl + col * LDS_STRIDE, half);
    }
    #pragma unroll
    for (int mt = 0; mt < 2; ++mt)
        #pragma unroll
        for (int nt = 0; nt < 2; ++nt) {
            acc[mt][nt] = wmma_f16(ahi[mt], bhi[nt], acc[mt][nt]);
            acc[mt][nt] = wmma_f16(ahi[mt], blo[nt], acc[mt][nt]);
            acc[mt][nt] = wmma_f16(alo[mt], bhi[nt], acc[mt][nt]);
        }
}

// ---------------------------------------------------------------------------
// Split-f16 GEMM:  C(MxN) = A(MxK) @ BT(NxK)^T, all operands hi/lo f16 planes.
// 256 threads = 8 waves; block tile 128(M) x 64(N); K step 32.
// EPI 0: sq_dist = max(sq[m]+sq[n]-2*acc,0)/sqrt(C) -> split planes outHi/Lo
// EPI 1: zT      = xT[m,n] + alpha*acc              -> split planes outHi/Lo
// EPI 2: out     = acc + bias[n]                    -> outf (f32)
// ---------------------------------------------------------------------------
template <int EPI>
__global__ __launch_bounds__(256) void gemm_planes(
    const _Float16* __restrict__ Ahi, const _Float16* __restrict__ Alo,
    const _Float16* __restrict__ BThi, const _Float16* __restrict__ BTlo,
    int M, int N, int K, int lda, int ldbt, int ldc,
    const float* __restrict__ sq,                 // EPI 0
    float* __restrict__ outf,                     // EPI 2
    _Float16* __restrict__ outHi, _Float16* __restrict__ outLo, // EPI 0/1
    const float* __restrict__ xres, int ldx,      // EPI 1 (xT residual)
    const float* __restrict__ bias)               // EPI 2
{
    __shared__ __align__(16) _Float16 sAhi[2][128 * LDS_STRIDE];
    __shared__ __align__(16) _Float16 sAlo[2][128 * LDS_STRIDE];
    __shared__ __align__(16) _Float16 sBhi[2][64 * LDS_STRIDE];
    __shared__ __align__(16) _Float16 sBlo[2][64 * LDS_STRIDE];

    const int t     = threadIdx.x;
    const int lane  = t & 31;
    const int wave  = t >> 5;
    const int waveM = wave & 3;
    const int waveN = wave >> 2;
    const int blockM = blockIdx.x * 128;
    const int blockN = blockIdx.y * 64;
    const int half  = lane >> 4;
    const int lrow  = lane & 15;

    const v8f zero8 = {0.f, 0.f, 0.f, 0.f, 0.f, 0.f, 0.f, 0.f};
    v8f acc[2][2];
    acc[0][0] = zero8; acc[0][1] = zero8; acc[1][0] = zero8; acc[1][1] = zero8;

#if USE_TDM
    // --- TDM double-buffered pipeline -------------------------------------
    const bool issuer = (t < 32);        // wave 0 drives the DMA
    const int  nsteps = K >> 5;
    if (issuer) {  // prologue: tile 0 -> buf 0
        tdm_load_2d(&sAhi[0][0], Ahi + (size_t)blockM * lda, 128, (unsigned)lda);
        tdm_load_2d(&sAlo[0][0], Alo + (size_t)blockM * lda, 128, (unsigned)lda);
        tdm_load_2d(&sBhi[0][0], BThi + (size_t)blockN * ldbt, 64, (unsigned)ldbt);
        tdm_load_2d(&sBlo[0][0], BTlo + (size_t)blockN * ldbt, 64, (unsigned)ldbt);
    }
    for (int s = 0; s < nsteps; ++s) {
        const int buf = s & 1;
        if (issuer) {
            if (s + 1 < nsteps) {  // issue tile s+1 into the other buffer
                const int k1 = (s + 1) << 5;
                tdm_load_2d(&sAhi[buf ^ 1][0], Ahi + (size_t)blockM * lda + k1, 128, (unsigned)lda);
                tdm_load_2d(&sAlo[buf ^ 1][0], Alo + (size_t)blockM * lda + k1, 128, (unsigned)lda);
                tdm_load_2d(&sBhi[buf ^ 1][0], BThi + (size_t)blockN * ldbt + k1, 64, (unsigned)ldbt);
                tdm_load_2d(&sBlo[buf ^ 1][0], BTlo + (size_t)blockN * ldbt + k1, 64, (unsigned)ldbt);
                __builtin_amdgcn_s_wait_tensorcnt((short)4); // tile s complete (in-order)
            } else {
                __builtin_amdgcn_s_wait_tensorcnt((short)0);
            }
        }
        __syncthreads();   // publish tile s to all waves
        mma_tile(&sAhi[buf][0], &sAlo[buf][0], &sBhi[buf][0], &sBlo[buf][0],
                 waveM, waveN, lrow, half, acc);
        __syncthreads();   // buffer reusable before tile s+2 is issued
    }
#else
    // --- fallback: VMEM staging into buffer 0 -----------------------------
    for (int k0 = 0; k0 < K; k0 += 32) {
        __syncthreads();
        #pragma unroll
        for (int i = 0; i < 2; ++i) {
            int v  = t + 256 * i;
            int r  = v >> 2;
            int c8 = (v & 3) << 3;
            size_t g = (size_t)(blockM + r) * lda + k0 + c8;
            *reinterpret_cast<uint4*>(&sAhi[0][r * LDS_STRIDE + c8]) =
                *reinterpret_cast<const uint4*>(Ahi + g);
            *reinterpret_cast<uint4*>(&sAlo[0][r * LDS_STRIDE + c8]) =
                *reinterpret_cast<const uint4*>(Alo + g);
        }
        {
            int r  = t >> 2;
            int c8 = (t & 3) << 3;
            size_t g = (size_t)(blockN + r) * ldbt + k0 + c8;
            *reinterpret_cast<uint4*>(&sBhi[0][r * LDS_STRIDE + c8]) =
                *reinterpret_cast<const uint4*>(BThi + g);
            *reinterpret_cast<uint4*>(&sBlo[0][r * LDS_STRIDE + c8]) =
                *reinterpret_cast<const uint4*>(BTlo + g);
        }
        __syncthreads();
        mma_tile(&sAhi[0][0], &sAlo[0][0], &sBhi[0][0], &sBlo[0][0],
                 waveM, waveN, lrow, half, acc);
    }
#endif

    // epilogue: C layout -> M = r + 8*half, N = lane&15
    #pragma unroll
    for (int mt = 0; mt < 2; ++mt)
        #pragma unroll
        for (int nt = 0; nt < 2; ++nt)
            #pragma unroll
            for (int r = 0; r < 8; ++r) {
                const int grow = blockM + waveM * 32 + mt * 16 + r + (half << 3);
                const int gcol = blockN + waveN * 32 + nt * 16 + lrow;
                const float a = acc[mt][nt][r];
                const size_t ci = (size_t)grow * ldc + gcol;
                if (EPI == 0) {
                    float sd = sq[grow] + sq[gcol] - 2.0f * a;
                    sd = fmaxf(sd, 0.0f) * RSQRT_C;
                    _Float16 h, l; split_f32(sd, h, l);
                    outHi[ci] = h; outLo[ci] = l;
                } else if (EPI == 1) {
                    float z = xres[(size_t)grow * ldx + gcol] + ALPHA * a;
                    _Float16 h, l; split_f32(z, h, l);
                    outHi[ci] = h; outLo[ci] = l;
                } else {
                    outf[ci] = a + bias[gcol];
                }
            }
}

// ---------------------------------------------------------------------------
// Helper kernels
// ---------------------------------------------------------------------------
__global__ void pack_x_kernel(const float* __restrict__ x,
                              _Float16* __restrict__ Xhi, _Float16* __restrict__ Xlo,
                              _Float16* __restrict__ XThi, _Float16* __restrict__ XTlo,
                              float* __restrict__ xT) {
    const size_t total = (size_t)BN * CN;
    for (size_t i = (size_t)blockIdx.x * blockDim.x + threadIdx.x; i < total;
         i += (size_t)gridDim.x * blockDim.x) {
        int r = (int)(i / CN), c = (int)(i % CN);
        float v = x[i];
        _Float16 h, l; split_f32(v, h, l);
        Xhi[i] = h; Xlo[i] = l;
        size_t j = (size_t)c * BN + r;
        XThi[j] = h; XTlo[j] = l; xT[j] = v;
    }
}

__global__ void pack_w_kernel(const float* __restrict__ W,
                              _Float16* __restrict__ WThi, _Float16* __restrict__ WTlo) {
    const size_t total = (size_t)CN * ON;
    for (size_t j = (size_t)blockIdx.x * blockDim.x + threadIdx.x; j < total;
         j += (size_t)gridDim.x * blockDim.x) {
        int o = (int)(j / CN), c = (int)(j % CN);
        _Float16 h, l; split_f32(W[(size_t)c * ON + o], h, l);
        WThi[j] = h; WTlo[j] = l;
    }
}

__global__ __launch_bounds__(256) void sqnorm_kernel(const float* __restrict__ x,
                                                     float* __restrict__ sq) {
    __shared__ float sm[256];
    const int row = blockIdx.x;
    float s = 0.f;
    for (int c = threadIdx.x; c < CN; c += 256) {
        float v = x[(size_t)row * CN + c];
        s += v * v;
    }
    sm[threadIdx.x] = s;
    __syncthreads();
    for (int st = 128; st > 0; st >>= 1) {
        if (threadIdx.x < st) sm[threadIdx.x] += sm[threadIdx.x + st];
        __syncthreads();
    }
    if (threadIdx.x == 0) sq[row] = sm[0];
}

__global__ __launch_bounds__(256) void stats_kernel(const _Float16* __restrict__ Dhi,
                                                    const _Float16* __restrict__ Dlo,
                                                    double* __restrict__ stats) {
    __shared__ double s0[256], s1[256];
    const int row = blockIdx.x;
    double a = 0.0, b = 0.0;
    for (int c = threadIdx.x; c < BN; c += 256) {
        if (c == row) continue;
        size_t i = (size_t)row * BN + c;
        double v = (double)((float)Dhi[i] + (float)Dlo[i]);
        a += v; b += v * v;
    }
    s0[threadIdx.x] = a; s1[threadIdx.x] = b;
    __syncthreads();
    for (int st = 128; st > 0; st >>= 1) {
        if (threadIdx.x < st) {
            s0[threadIdx.x] += s0[threadIdx.x + st];
            s1[threadIdx.x] += s1[threadIdx.x + st];
        }
        __syncthreads();
    }
    if (threadIdx.x == 0) {
        atomicAdd(&stats[0], s0[0]);
        atomicAdd(&stats[1], s1[0]);
    }
}

__global__ __launch_bounds__(256) void weights_kernel(_Float16* __restrict__ Dhi,
                                                      _Float16* __restrict__ Dlo,
                                                      const double* __restrict__ stats) {
    const double cnt  = (double)BN * (double)(BN - 1);
    const double mean = stats[0] / cnt;
    const double var  = (stats[1] - cnt * mean * mean) / (cnt - 1.0);
    const float rstd  = (float)(1.0 / sqrt(var));
    const int row = blockIdx.x;
    for (int c = threadIdx.x; c < BN; c += 256) {
        size_t i = (size_t)row * BN + c;
        float w;
        if (c == row) w = 0.0f;
        else {
            float sd = (float)Dhi[i] + (float)Dlo[i];
            w = __expf(-(sd * rstd) * RBF);
        }
        _Float16 h, l; split_f32(w, h, l);
        Dhi[i] = h; Dlo[i] = l;
    }
}

__global__ __launch_bounds__(256) void rowsum_kernel(const _Float16* __restrict__ Whi,
                                                     const _Float16* __restrict__ Wlo,
                                                     float* __restrict__ dinv) {
    __shared__ float sm[256];
    const int row = blockIdx.x;
    float s = 0.f;
    for (int c = threadIdx.x; c < BN; c += 256) {
        size_t i = (size_t)row * BN + c;
        s += (float)Whi[i] + (float)Wlo[i];
    }
    sm[threadIdx.x] = s;
    __syncthreads();
    for (int st = 128; st > 0; st >>= 1) {
        if (threadIdx.x < st) sm[threadIdx.x] += sm[threadIdx.x + st];
        __syncthreads();
    }
    if (threadIdx.x == 0) dinv[row] = rsqrtf(1e-4f + sm[0]);
}

__global__ __launch_bounds__(256) void normpack_kernel(_Float16* __restrict__ Dhi,
                                                       _Float16* __restrict__ Dlo,
                                                       const float* __restrict__ dinv) {
    const int row = blockIdx.x;
    const float dr = dinv[row];
    for (int c = threadIdx.x; c < BN; c += 256) {
        size_t i = (size_t)row * BN + c;
        float v = ((float)Dhi[i] + (float)Dlo[i]) * dr * dinv[c];
        _Float16 h, l; split_f32(v, h, l);
        Dhi[i] = h; Dlo[i] = l;
    }
}

__global__ __launch_bounds__(256) void transpose_planes(
    const _Float16* __restrict__ shi, const _Float16* __restrict__ slo,
    _Float16* __restrict__ dhi, _Float16* __restrict__ dlo, int R, int C) {
    __shared__ _Float16 th[32][33], tl[32][33];
    const int cb = blockIdx.x * 32, rb = blockIdx.y * 32;
    const int tx = threadIdx.x & 31, ty = threadIdx.x >> 5;
    for (int i = ty; i < 32; i += 8) {
        size_t g = (size_t)(rb + i) * C + cb + tx;
        th[i][tx] = shi[g];
        tl[i][tx] = slo[g];
    }
    __syncthreads();
    for (int i = ty; i < 32; i += 8) {
        size_t g = (size_t)(cb + i) * R + rb + tx;
        dhi[g] = th[tx][i];
        dlo[g] = tl[tx][i];
    }
}

// ---------------------------------------------------------------------------
extern "C" void kernel_launch(void* const* d_in, const int* in_sizes, int n_in,
                              void* d_out, int out_size, void* d_ws, size_t ws_size,
                              hipStream_t stream) {
    const float* x    = (const float*)d_in[0];   // [4096,512]
    const float* W    = (const float*)d_in[1];   // [512,512]
    const float* bias = (const float*)d_in[2];   // [512]
    float* out = (float*)d_out;                  // [4096,512]

    size_t off = 0;
    auto ws_alloc = [&](size_t bytes) {
        void* p = (char*)d_ws + off;
        off += (bytes + 255) & ~(size_t)255;
        return p;
    };
    const size_t XE = (size_t)BN * CN;
    const size_t SE = (size_t)BN * BN;
    _Float16* Xhi  = (_Float16*)ws_alloc(XE * 2);
    _Float16* Xlo  = (_Float16*)ws_alloc(XE * 2);
    _Float16* XThi = (_Float16*)ws_alloc(XE * 2);   // = z_0^T planes
    _Float16* XTlo = (_Float16*)ws_alloc(XE * 2);
    float*    xT   = (float*)   ws_alloc(XE * 4);   // residual, [CN][BN]
    _Float16* WThi = (_Float16*)ws_alloc((size_t)CN * ON * 2);
    _Float16* WTlo = (_Float16*)ws_alloc((size_t)CN * ON * 2);
    _Float16* Shi  = (_Float16*)ws_alloc(SE * 2);   // sq_dist -> w -> S
    _Float16* Slo  = (_Float16*)ws_alloc(SE * 2);
    _Float16* ZT1h = (_Float16*)ws_alloc(XE * 2);
    _Float16* ZT1l = (_Float16*)ws_alloc(XE * 2);
    _Float16* ZT2h = (_Float16*)ws_alloc(XE * 2);
    _Float16* ZT2l = (_Float16*)ws_alloc(XE * 2);
    _Float16* Zfh  = (_Float16*)ws_alloc(XE * 2);   // z (non-transposed)
    _Float16* Zfl  = (_Float16*)ws_alloc(XE * 2);
    float*    sq   = (float*)   ws_alloc((size_t)BN * 4);
    float*    dinv = (float*)   ws_alloc((size_t)BN * 4);
    double*   stats= (double*)  ws_alloc(2 * sizeof(double));

    (void)hipMemsetAsync(stats, 0, 2 * sizeof(double), stream);

    pack_x_kernel<<<2048, 256, 0, stream>>>(x, Xhi, Xlo, XThi, XTlo, xT);
    pack_w_kernel<<<1024, 256, 0, stream>>>(W, WThi, WTlo);
    sqnorm_kernel<<<BN, 256, 0, stream>>>(x, sq);

    // Gram + fused sq_dist (BT operand of X@X^T is X itself)
    gemm_planes<0><<<dim3(BN / 128, BN / 64), 256, 0, stream>>>(
        Xhi, Xlo, Xhi, Xlo, BN, BN, CN, CN, CN, BN,
        sq, nullptr, Shi, Slo, nullptr, 0, nullptr);

    stats_kernel<<<BN, 256, 0, stream>>>(Shi, Slo, stats);
    weights_kernel<<<BN, 256, 0, stream>>>(Shi, Slo, stats);
    rowsum_kernel<<<BN, 256, 0, stream>>>(Shi, Slo, dinv);
    normpack_kernel<<<BN, 256, 0, stream>>>(Shi, Slo, dinv);

    // Neumann in transposed space: zT_{k+1} = xT + alpha * zT_k @ S (S = S^T)
    const _Float16* zinh = XThi; const _Float16* zinl = XTlo;
    for (int it = 0; it < NEUMANN_ITERS; ++it) {
        _Float16* zoh = (it & 1) ? ZT2h : ZT1h;
        _Float16* zol = (it & 1) ? ZT2l : ZT1l;
        gemm_planes<1><<<dim3(CN / 128, BN / 64), 256, 0, stream>>>(
            zinh, zinl, Shi, Slo, CN, BN, BN, BN, BN, BN,
            nullptr, nullptr, zoh, zol, xT, BN, nullptr);
        zinh = zoh; zinl = zol;
    }

    // zT -> z, then out = z @ W + bias  (BT operand = W^T planes)
    transpose_planes<<<dim3(BN / 32, CN / 32), 256, 0, stream>>>(
        zinh, zinl, Zfh, Zfl, CN, BN);
    gemm_planes<2><<<dim3(BN / 128, ON / 64), 256, 0, stream>>>(
        Zfh, Zfl, WThi, WTlo, BN, ON, CN, CN, CN, ON,
        nullptr, out, nullptr, nullptr, nullptr, 0, bias);
}
